// MixtureofExpertsBlock_56564719289045
// MI455X (gfx1250) — matrix operand order repaired and checked by reading, not compiled, not run on hardware
//
#include <hip/hip_runtime.h>
#include <hip/hip_bf16.h>
#include <math.h>

// ---------------------------------------------------------------------------
// MoE block for MI455X (gfx1250): routed grouped-GEMM with bf16 WMMA,
// async global->LDS A-tile staging (ASYNCcnt), double buffered, 64-wide K stages.
// ---------------------------------------------------------------------------

typedef __attribute__((ext_vector_type(16))) __bf16 v16bf;
typedef __attribute__((ext_vector_type(8)))  __bf16 v8bf;
typedef __attribute__((ext_vector_type(8)))  float  v8f;

union BFrag { v16bf v; v8bf h[2]; };

#define KTOP 2
#define KSTAGE 64     // K elements staged per buffer (2 WMMA sub-steps)
#define A_STRIDE 72   // bf16 elements per staged A row (144B = 9x16B, bank-skewed)

// Async copy global->LDS, offset applied to both addresses, 16B per lane (ASYNCcnt).
#define ASYNC_LD_B128(OFFSTR, LDS, GADDR)                                      \
    asm volatile("global_load_async_to_lds_b128 %0, %1, off offset:" OFFSTR    \
                 :: "v"(LDS), "v"(GADDR) : "memory")

__device__ __forceinline__ void stage_row_128B(uint32_t lds, const void* g) {
    ASYNC_LD_B128("0",  lds, g);
    ASYNC_LD_B128("32", lds, g);
    ASYNC_LD_B128("64", lds, g);
    ASYNC_LD_B128("96", lds, g);
}
__device__ __forceinline__ void wait_async0() {
#if __has_builtin(__builtin_amdgcn_s_wait_asynccnt)
    __builtin_amdgcn_s_wait_asynccnt(0);
#else
    asm volatile("s_wait_asynccnt 0x0" ::: "memory");
#endif
}

// ---------------- zero out + counters ----------------
__global__ void moe_zero_kernel(float* __restrict__ out, size_t n,
                                int* __restrict__ counts, int numExperts) {
    size_t i = (size_t)blockIdx.x * blockDim.x + threadIdx.x;
    size_t stride = (size_t)gridDim.x * blockDim.x;
    for (size_t j = i; j < n; j += stride) out[j] = 0.0f;
    if (i < (size_t)numExperts) counts[i] = 0;
}

// ---------------- f32 -> bf16 conversion ----------------
__global__ void moe_cvt_bf16_kernel(const float* __restrict__ in,
                                    __bf16* __restrict__ out, size_t n) {
    size_t i = (size_t)blockIdx.x * blockDim.x + threadIdx.x;
    size_t stride = (size_t)gridDim.x * blockDim.x;
    for (size_t j = i; j < n; j += stride) out[j] = (__bf16)in[j];
}

// ---------------- router: logits, top-2 softmax, expert lists ----------------
__global__ __launch_bounds__(256)
void moe_router_kernel(const float* __restrict__ x, const float* __restrict__ wr,
                       int N, int D, int E, int cap,
                       int* __restrict__ counts, int* __restrict__ tokens,
                       float* __restrict__ wts) {
    __shared__ float lg[32][8];
    int tid = threadIdx.x;
    int tl = tid >> 3, e = tid & 7;
    int token = blockIdx.x * 32 + tl;
    if (token < N && e < E) {
        const float4* xr = reinterpret_cast<const float4*>(x + (size_t)token * D);
        const float4* wrow = reinterpret_cast<const float4*>(wr + (size_t)e * D);
        float acc = 0.0f;
        int n4 = D >> 2;
        for (int i = 0; i < n4; ++i) {
            float4 a = xr[i], b = wrow[i];
            acc += a.x * b.x + a.y * b.y + a.z * b.z + a.w * b.w;
        }
        lg[tl][e] = acc;
    }
    __syncthreads();
    if (tid < 32) {
        int tok2 = blockIdx.x * 32 + tid;
        if (tok2 < N) {
            float l0 = -INFINITY, l1 = -INFINITY;
            int i0 = 0, i1 = 0;
            for (int j = 0; j < E; ++j) {
                float v = lg[tid][j];
                if (v > l0)      { l1 = l0; i1 = i0; l0 = v; i0 = j; }
                else if (v > l1) { l1 = v;  i1 = j; }
            }
            float w0 = 1.0f / (1.0f + __expf(l1 - l0));   // softmax over top-2
            float w1 = 1.0f - w0;
            int p0 = atomicAdd(&counts[i0], 1);
            tokens[(size_t)i0 * cap + p0] = tok2;
            wts   [(size_t)i0 * cap + p0] = w0;
            int p1 = atomicAdd(&counts[i1], 1);
            tokens[(size_t)i1 * cap + p1] = tok2;
            wts   [(size_t)i1 * cap + p1] = w1;
        }
    }
}

// ---------------- 16-row padded prefix offsets ----------------
__global__ void moe_prefix_kernel(const int* __restrict__ counts,
                                  int* __restrict__ offsets, int E) {
    if (threadIdx.x == 0 && blockIdx.x == 0) {
        int o = 0;
        for (int e = 0; e < E; ++e) {
            offsets[e] = o;
            o += (counts[e] + 15) & ~15;
        }
        offsets[E] = o;
    }
}

// ---------------- up-projection grouped GEMM + GELU (bf16 WMMA) ----------------
// C[m,n] = sum_k x[token[m],k] * w_up[e,n,k];  h = gelu(C + b_up)
template<int TPW>
__global__ __launch_bounds__(256)
void moe_up_kernel(const __bf16* __restrict__ xb, const __bf16* __restrict__ wub,
                   const float* __restrict__ b_up,
                   const int* __restrict__ tokens, const int* __restrict__ counts,
                   const int* __restrict__ offsets,
                   __bf16* __restrict__ hbuf,
                   int cap, int D, int U) {
    __shared__ __bf16 aSh[2][16 * A_STRIDE];

    int e = blockIdx.z;
    int count = counts[e];
    int rowStart = blockIdx.y * 16;
    if (rowStart >= count) return;                  // past this expert's tokens

    int wave = threadIdx.x >> 5;
    int lane = threadIdx.x & 31;
    int half = lane >> 4;
    int ln   = lane & 15;
    int nBase = blockIdx.x * (8 * TPW * 16) + wave * (TPW * 16);

    // per-lane gathered A row (row = ln), clamped on ragged tail
    int arow = rowStart + ln; if (arow > count - 1) arow = count - 1;
    int tok = tokens[(size_t)e * cap + arow];
    const __bf16* aRow = xb + (size_t)tok * D;      // global source for staging
    const __bf16* wrow = wub + (size_t)e * U * D;

    // staging lane map: row = ln, first 16B chunk = half, offsets cover the rest
    uint32_t ldsRow = (uint32_t)(ln * A_STRIDE + half * 8) * 2u;  // bytes
    uint32_t ldsB0 = (uint32_t)(uintptr_t)&aSh[0][0];
    uint32_t ldsB1 = (uint32_t)(uintptr_t)&aSh[1][0];

    if (wave == 0) {                                 // preload stage 0
        stage_row_128B(ldsB0 + ldsRow, aRow + half * 8);
        wait_async0();
    }
    __syncthreads();

    v8f acc[TPW] = {};
    int buf = 0;

    for (int k0 = 0; k0 < D; k0 += KSTAGE) {
        if (wave == 0 && k0 + KSTAGE < D)
            stage_row_128B((buf ? ldsB0 : ldsB1) + ldsRow,
                           aRow + k0 + KSTAGE + half * 8);
        const __bf16* as = &aSh[buf][ln * A_STRIDE];
#pragma unroll
        for (int ks = 0; ks < KSTAGE; ks += 32) {
            BFrag fa;
            fa.h[0] = *reinterpret_cast<const v8bf*>(as + ks + half * 8);       // ds_load_b128
            fa.h[1] = *reinterpret_cast<const v8bf*>(as + ks + 16 + half * 8);
#pragma unroll
            for (int t = 0; t < TPW; ++t) {
                int n = nBase + t * 16 + ln;
                const __bf16* bp = wrow + (size_t)n * D + k0 + ks + half * 8;
                BFrag fb;
                fb.h[0] = *reinterpret_cast<const v8bf*>(bp);
                fb.h[1] = *reinterpret_cast<const v8bf*>(bp + 16);
                acc[t] = __builtin_amdgcn_wmma_f32_16x16x32_bf16(
                    false, fa.v, false, fb.v, (short)0, acc[t], false, false);
            }
        }
        if (wave == 0) wait_async0();
        __syncthreads();
        buf ^= 1;
    }

    size_t hRowBase = (size_t)offsets[e] + rowStart;
#pragma unroll
    for (int t = 0; t < TPW; ++t) {
        int n = nBase + t * 16 + ln;
        float bias = b_up[(size_t)e * U + n];
#pragma unroll
        for (int r = 0; r < 8; ++r) {
            float v = acc[t][r] + bias;
            v = 0.5f * v * (1.0f + erff(v * 0.70710678118f));   // exact GELU
            hbuf[(hRowBase + (size_t)(r + 8 * half)) * U + n] = (__bf16)v;
        }
    }
}

// ---------------- down-projection grouped GEMM + weighted scatter ----------------
// out[token,d] += w_router * (sum_u h[m,u]*w_down[e,d,u] + b_down[e,d])
template<int TPW>
__global__ __launch_bounds__(256)
void moe_down_kernel(const __bf16* __restrict__ hbuf, const __bf16* __restrict__ wdb,
                     const float* __restrict__ b_down,
                     const int* __restrict__ tokens, const float* __restrict__ wts,
                     const int* __restrict__ counts, const int* __restrict__ offsets,
                     float* __restrict__ out,
                     int cap, int D, int U) {
    __shared__ __bf16 aSh[2][16 * A_STRIDE];

    int e = blockIdx.z;
    int count = counts[e];
    int rowStart = blockIdx.y * 16;
    if (rowStart >= count) return;

    int wave = threadIdx.x >> 5;
    int lane = threadIdx.x & 31;
    int half = lane >> 4;
    int ln   = lane & 15;
    int nBase = blockIdx.x * (8 * TPW * 16) + wave * (TPW * 16);

    size_t hRowBase = (size_t)offsets[e] + rowStart;
    const __bf16* aRow = hbuf + (hRowBase + (size_t)ln) * U;   // padded region valid
    const __bf16* wrow = wdb + (size_t)e * (size_t)D * U;

    uint32_t ldsRow = (uint32_t)(ln * A_STRIDE + half * 8) * 2u;
    uint32_t ldsB0 = (uint32_t)(uintptr_t)&aSh[0][0];
    uint32_t ldsB1 = (uint32_t)(uintptr_t)&aSh[1][0];

    if (wave == 0) {
        stage_row_128B(ldsB0 + ldsRow, aRow + half * 8);
        wait_async0();
    }
    __syncthreads();

    v8f acc[TPW] = {};
    int buf = 0;

    for (int k0 = 0; k0 < U; k0 += KSTAGE) {
        if (wave == 0 && k0 + KSTAGE < U)
            stage_row_128B((buf ? ldsB0 : ldsB1) + ldsRow,
                           aRow + k0 + KSTAGE + half * 8);
        const __bf16* as = &aSh[buf][ln * A_STRIDE];
#pragma unroll
        for (int ks = 0; ks < KSTAGE; ks += 32) {
            BFrag fa;
            fa.h[0] = *reinterpret_cast<const v8bf*>(as + ks + half * 8);
            fa.h[1] = *reinterpret_cast<const v8bf*>(as + ks + 16 + half * 8);
#pragma unroll
            for (int t = 0; t < TPW; ++t) {
                int n = nBase + t * 16 + ln;
                const __bf16* bp = wrow + (size_t)n * U + k0 + ks + half * 8;
                BFrag fb;
                fb.h[0] = *reinterpret_cast<const v8bf*>(bp);
                fb.h[1] = *reinterpret_cast<const v8bf*>(bp + 16);
                acc[t] = __builtin_amdgcn_wmma_f32_16x16x32_bf16(
                    false, fa.v, false, fb.v, (short)0, acc[t], false, false);
            }
        }
        if (wave == 0) wait_async0();
        __syncthreads();
        buf ^= 1;
    }

    int tokR[8]; float wtR[8];
#pragma unroll
    for (int r = 0; r < 8; ++r) {
        int row = rowStart + r + 8 * half;
        bool valid = row < count;
        tokR[r] = valid ? tokens[(size_t)e * cap + row] : -1;
        wtR[r]  = valid ? wts[(size_t)e * cap + row]    : 0.0f;
    }
#pragma unroll
    for (int t = 0; t < TPW; ++t) {
        int d = nBase + t * 16 + ln;
        float bias = b_down[(size_t)e * D + d];
#pragma unroll
        for (int r = 0; r < 8; ++r) {
            if (tokR[r] >= 0)
                atomicAdd(out + (size_t)tokR[r] * D + d, wtR[r] * (acc[t][r] + bias));
        }
    }
}

// ---------------------------------------------------------------------------
extern "C" void kernel_launch(void* const* d_in, const int* in_sizes, int n_in,
                              void* d_out, int out_size, void* d_ws, size_t ws_size,
                              hipStream_t stream) {
    const float* x        = (const float*)d_in[0];
    const float* w_router = (const float*)d_in[1];
    const float* w_up     = (const float*)d_in[2];
    const float* b_up     = (const float*)d_in[3];
    const float* w_down   = (const float*)d_in[4];
    const float* b_down   = (const float*)d_in[5];
    float* out = (float*)d_out;

    const int D = in_sizes[2] / in_sizes[3];   // (E*U*D)/(E*U)
    const int U = in_sizes[4] / in_sizes[5];   // (E*D*U)/(E*D)
    const int E = in_sizes[1] / D;
    const int N = in_sizes[0] / D;             // B*S tokens

    // ---- carve workspace (256B aligned) ----
    char* base = (char*)d_ws;
    size_t off = 0;
    auto carve = [&](size_t bytes) -> char* {
        off = (off + 255) & ~(size_t)255;
        char* p = base + off;
        off += bytes;
        return p;
    };
    __bf16* xb   = (__bf16*)carve((size_t)N * D * 2);
    __bf16* wub  = (__bf16*)carve((size_t)E * U * D * 2);
    __bf16* wdb  = (__bf16*)carve((size_t)E * D * U * 2);
    __bf16* hb   = (__bf16*)carve(((size_t)N * KTOP + 16u * E) * U * 2);
    int*   tokens  = (int*)carve((size_t)E * N * 4);
    float* wts     = (float*)carve((size_t)E * N * 4);
    int*   counts  = (int*)carve((size_t)E * 4);
    int*   offsets = (int*)carve((size_t)(E + 1) * 4);
    (void)ws_size; (void)n_in; (void)out_size;

    // 1. zero output + expert counters
    moe_zero_kernel<<<1024, 256, 0, stream>>>(out, (size_t)N * D, counts, E);
    // 2. f32 -> bf16 operand conversion
    moe_cvt_bf16_kernel<<<2048, 256, 0, stream>>>(x, xb, (size_t)N * D);
    moe_cvt_bf16_kernel<<<4096, 256, 0, stream>>>(w_up, wub, (size_t)E * U * D);
    moe_cvt_bf16_kernel<<<4096, 256, 0, stream>>>(w_down, wdb, (size_t)E * D * U);
    // 3. router: logits -> top-2 softmax -> expert token lists
    moe_router_kernel<<<(N + 31) / 32, 256, 0, stream>>>(x, w_router, N, D, E, N,
                                                         counts, tokens, wts);
    // 4. padded prefix offsets into the compact h buffer
    moe_prefix_kernel<<<1, 32, 0, stream>>>(counts, offsets, E);
    // 5. up-projection grouped GEMM + GELU (worst-case grid, early exit)
    int maxMT = (N + 15) / 16;
    dim3 gUp(U / (8 * 4 * 16), maxMT, E);     // 8 waves x 4 tiles = 512 cols/block
    moe_up_kernel<4><<<gUp, 256, 0, stream>>>(xb, wub, b_up, tokens, counts,
                                              offsets, hb, N, D, U);
    // 6. down-projection grouped GEMM + weighted atomic scatter
    dim3 gDn(D / (8 * 4 * 16), maxMT, E);     // 8 waves x 4 tiles = 512 cols/block
    moe_down_kernel<4><<<gDn, 256, 0, stream>>>(hb, wdb, b_down, tokens, wts,
                                                counts, offsets, out, N, D, U);
}
